// GATEncoder_541165879950
// MI455X (gfx1250) — matrix operand order repaired and checked by reading, not compiled, not run on hardware
//
#include <hip/hip_runtime.h>
#include <hip/hip_bf16.h>

// ---------------- CDNA5 WMMA types ----------------
typedef __attribute__((ext_vector_type(16))) __bf16 v16bf;
typedef __attribute__((ext_vector_type(8)))  __bf16 v8bf;
typedef __attribute__((ext_vector_type(8)))  float  v8f;

union ABu { v16bf v; v8bf h[2]; };

#define WMMA_BF16(A, Bm, C) \
  __builtin_amdgcn_wmma_f32_16x16x32_bf16(false, (A), false, (Bm), (short)0, (C), false, false)

static constexpr int Bb = 8;
static constexpr int S  = 2048;
static constexpr int D  = 256;

__device__ __forceinline__ unsigned short f2bf(float f) {
  unsigned u = __float_as_uint(f);
  u += 0x7FFFu + ((u >> 16) & 1u);   // round-to-nearest-even
  return (unsigned short)(u >> 16);
}

// ---------------- small utility kernels ----------------
__global__ void k_zero(float* p, int n) {
  int i = blockIdx.x * 256 + threadIdx.x;
  if (i < n) p[i] = 0.f;
}

__global__ void k_cvt(const float* __restrict__ x, unsigned short* __restrict__ y, int n) {
  int i = blockIdx.x * 256 + threadIdx.x;
  if (i < n) y[i] = f2bf(x[i]);
}

// Wt[d][k] = bf16(W[k][d]) : 256x256, coalesced writes
__global__ void k_cvtW(const float* __restrict__ W, unsigned short* __restrict__ Wt) {
  int i = blockIdx.x * 256 + threadIdx.x;  // over 65536
  int d = i >> 8, k = i & 255;
  Wt[d * 256 + k] = f2bf(W[k * 256 + d]);
}

// denom[b][s] = rowsum(adj)[s] + colsum(adj)[s]  (64x64 tiles, atomics)
__global__ void k_denom(const float* __restrict__ adj, float* __restrict__ denom) {
  int b = blockIdx.z, i0 = blockIdx.y * 64, j0 = blockIdx.x * 64;
  const float* A = adj + (size_t)b * S * S;
  __shared__ float rowacc[64];
  __shared__ float colacc[64];
  int tid = threadIdx.x, lane = tid & 31;
  if (tid < 64) { rowacc[tid] = 0.f; colacc[tid] = 0.f; }
  __syncthreads();
  int tx = tid & 63, ty = tid >> 6;     // 64 cols x 4 row-groups
  float colpart = 0.f;
  for (int r = 0; r < 16; r++) {
    int row = ty * 16 + r;
    float v = A[(size_t)(i0 + row) * S + j0 + tx];
    colpart += v;
    float s = v;                         // reduce across the wave's 32 columns
    s += __shfl_xor(s, 16); s += __shfl_xor(s, 8); s += __shfl_xor(s, 4);
    s += __shfl_xor(s, 2);  s += __shfl_xor(s, 1);
    if (lane == 0) atomicAdd(&rowacc[row], s);
  }
  atomicAdd(&colacc[tx], colpart);
  __syncthreads();
  if (tid < 64) {
    atomicAdd(&denom[(size_t)b * S + i0 + tid], rowacc[tid]);
    atomicAdd(&denom[(size_t)b * S + j0 + tid], colacc[tid]);
  }
}

// ---------------- Yt = (Xb @ W)^T in bf16, WMMA ----------------
__global__ void k_gemmY(const unsigned short* __restrict__ Xb,
                        const unsigned short* __restrict__ Wt,
                        unsigned short* __restrict__ Yt) {
  int b = blockIdx.y, m0 = blockIdx.x * 64;
  int tid = threadIdx.x, lane = tid & 31, w = tid >> 5;
  int hl = lane >> 4, nl = lane & 15;
  const unsigned short* Xrow = Xb + (size_t)b * S * D;
  v8f acc[4][2] = {};
  for (int kk = 0; kk < D; kk += 32) {
    ABu a[4];
#pragma unroll
    for (int mi = 0; mi < 4; mi++) {
      const unsigned short* p = Xrow + (size_t)(m0 + mi * 16 + nl) * D + kk;
      a[mi].h[0] = *(const v8bf*)(p + hl * 16);
      a[mi].h[1] = *(const v8bf*)(p + hl * 16 + 8);
    }
#pragma unroll
    for (int ni = 0; ni < 2; ni++) {
      int d = w * 32 + ni * 16 + nl;
      v16bf bfr = *(const v16bf*)(Wt + (size_t)d * 256 + kk + hl * 16);
#pragma unroll
      for (int mi = 0; mi < 4; mi++)
        acc[mi][ni] = WMMA_BF16(a[mi].v, bfr, acc[mi][ni]);
    }
  }
  for (int mi = 0; mi < 4; mi++)
    for (int ni = 0; ni < 2; ni++) {
      int d = w * 32 + ni * 16 + nl;
      unsigned short* orow = Yt + ((size_t)b * D + d) * S;
#pragma unroll
      for (int r = 0; r < 8; r++)
        orow[m0 + mi * 16 + r + 8 * hl] = f2bf(acc[mi][ni][r]);
    }
}

// ---------------- fused GAT attention + epilogue ----------------
// block: (b, 64-row tile). 8 waves. Streams t in 64-col tiles.
__global__ __launch_bounds__(256)
void k_attn(const float* __restrict__ X, const unsigned short* __restrict__ Xb,
            const unsigned short* __restrict__ Yt, const float* __restrict__ adj,
            const float* __restrict__ denom, const float* __restrict__ bias,
            const float* __restrict__ gamma, const float* __restrict__ beta,
            float* __restrict__ out) {
  int b = blockIdx.y, m0 = blockIdx.x * 64;
  int tid = threadIdx.x, lane = tid & 31, w = tid >> 5;
  int hl = lane >> 4, nl = lane & 15;
  int mi = w & 3, tjp = w >> 2;          // score-tile assignment

  __shared__ unsigned short P[64][72];   // bf16 probabilities (144B rows, 16B-aligned)
  __shared__ float At[64][68];           // transposed adj tile (272B rows, 16B-aligned)
  __shared__ float lsum[64];             // exp row sums
  __shared__ float rden[64];             // 1/denom per row
  __shared__ float s1a[64], s2a[64];     // LN partial sums

  const float* Ab = adj + (size_t)b * S * S;
  const unsigned short* Xbb = Xb + (size_t)b * S * D;
  const float* Xf = X + (size_t)b * S * D;

  if (tid < 64) {
    float dv = denom[(size_t)b * S + m0 + tid];
    if (dv == 0.f) dv = 1.f;
    rden[tid] = 1.f / dv;
    lsum[tid] = 0.f; s1a[tid] = 0.f; s2a[tid] = 0.f;
  }
  __syncthreads();

  // per-lane constants for the async adj-tile fill (wave w owns rows [8w, 8w+8))
  int arow0 = w * 8 + hl;                // + 2*i per issue
  int acol  = nl * 4;                    // 16-byte column chunk
  unsigned lds_at0 = (unsigned)(size_t)(&At[0][0]);

  v8f acc[4][2] = {};                    // out tile: 64 rows x (wave's 32 cols)

  for (int t0 = 0; t0 < S; t0 += 64) {
    // ---- async fill of transposed adj tile: At[tt][mm] = adj[t0+tt][m0+mm] ----
    // GLOBAL_LOAD_ASYNC_TO_LDS_B128: LDS gets data without a VGPR round trip.
#pragma unroll
    for (int i = 0; i < 4; i++) {
      int trow = arow0 + 2 * i;
      unsigned long long ga =
          (unsigned long long)(Ab + (size_t)(t0 + trow) * S + m0 + acol);
      unsigned la = lds_at0 + (unsigned)(trow * 68 + acol) * 4u;
      asm volatile("global_load_async_to_lds_b128 %0, %1, off"
                   :: "v"(la), "v"(ga) : "memory");
    }

    // ---- phase 1: score tiles  S = Xm * Xt^T  (each wave: 2 C-tiles) ----
    v8f c[2] = {};
    for (int kk = 0; kk < D; kk += 32) {
      ABu a;
      const unsigned short* p = Xbb + (size_t)(m0 + mi * 16 + nl) * D + kk;
      a.h[0] = *(const v8bf*)(p + hl * 16);
      a.h[1] = *(const v8bf*)(p + hl * 16 + 8);
#pragma unroll
      for (int j = 0; j < 2; j++) {
        int t = t0 + (tjp * 2 + j) * 16 + nl;
        v16bf bfr = *(const v16bf*)(Xbb + (size_t)t * D + kk + hl * 16);
        c[j] = WMMA_BF16(a.v, bfr, c[j]);
      }
    }
    asm volatile("s_wait_asynccnt 0x0" ::: "memory");   // our At writes landed
    __syncthreads();                      // all waves' At visible; prev P-reads done

    // ---- exp * normalized adj; online row-sum; P -> LDS bf16 ----
#pragma unroll
    for (int j = 0; j < 2; j++) {
      int colL = (tjp * 2 + j) * 16 + nl;
#pragma unroll
      for (int r = 0; r < 8; r++) {
        int rowL = mi * 16 + r + 8 * hl;
        float av = (Ab[(size_t)(m0 + rowL) * S + t0 + colL] + At[colL][rowL]) * rden[rowL];
        float e = __expf(c[j][r] * 0.0625f) * av;   // 1/sqrt(256)
        float s = e;
        s += __shfl_xor(s, 8); s += __shfl_xor(s, 4);
        s += __shfl_xor(s, 2); s += __shfl_xor(s, 1);
        if (nl == 0) atomicAdd(&lsum[rowL], s);
        P[rowL][colL] = f2bf(e);
      }
    }
    __syncthreads();                      // P ready

    // ---- phase 2: acc += P[64x64] @ Yt-chunk (wave's 32 cols) ----
#pragma unroll
    for (int kk2 = 0; kk2 < 64; kk2 += 32) {
      ABu pa[4];
#pragma unroll
      for (int m2 = 0; m2 < 4; m2++) {
        const unsigned short* pp = &P[m2 * 16 + nl][kk2];
        pa[m2].h[0] = *(const v8bf*)(pp + hl * 16);
        pa[m2].h[1] = *(const v8bf*)(pp + hl * 16 + 8);
      }
#pragma unroll
      for (int ni = 0; ni < 2; ni++) {
        int d = w * 32 + ni * 16 + nl;
        v16bf bfr = *(const v16bf*)(Yt + ((size_t)b * D + d) * S + t0 + kk2 + hl * 16);
#pragma unroll
        for (int m2 = 0; m2 < 4; m2++)
          acc[m2][ni] = WMMA_BF16(pa[m2].v, bfr, acc[m2][ni]);
      }
    }
    __syncthreads();                      // safe to overwrite At/P next iter
  }

  // ---- epilogue: normalize, bias, leaky-relu, residual (in registers) ----
  for (int m2 = 0; m2 < 4; m2++)
    for (int ni = 0; ni < 2; ni++) {
      int col = w * 32 + ni * 16 + nl;
      float bv = bias[col];
#pragma unroll
      for (int r = 0; r < 8; r++) {
        int rowL = m2 * 16 + r + 8 * hl;
        float o = acc[m2][ni][r] / (lsum[rowL] + 1e-10f) + bv;
        o = (o > 0.f) ? o : 0.01f * o;
        acc[m2][ni][r] = Xf[(size_t)(m0 + rowL) * D + col] + o;
      }
    }
  // LayerNorm stats: each wave reduces its 32 cols, atomically merge
  for (int m2 = 0; m2 < 4; m2++) {
#pragma unroll
    for (int r = 0; r < 8; r++) {
      int rowL = m2 * 16 + r + 8 * hl;
      float h0 = acc[m2][0][r], h1 = acc[m2][1][r];
      float p1 = h0 + h1, p2 = h0 * h0 + h1 * h1;
      p1 += __shfl_xor(p1, 8); p1 += __shfl_xor(p1, 4);
      p1 += __shfl_xor(p1, 2); p1 += __shfl_xor(p1, 1);
      p2 += __shfl_xor(p2, 8); p2 += __shfl_xor(p2, 4);
      p2 += __shfl_xor(p2, 2); p2 += __shfl_xor(p2, 1);
      if (nl == 0) { atomicAdd(&s1a[rowL], p1); atomicAdd(&s2a[rowL], p2); }
    }
  }
  __syncthreads();
  for (int m2 = 0; m2 < 4; m2++)
    for (int ni = 0; ni < 2; ni++) {
      int col = w * 32 + ni * 16 + nl;
      float g = gamma[col], be = beta[col];
#pragma unroll
      for (int r = 0; r < 8; r++) {
        int rowL = m2 * 16 + r + 8 * hl;
        float mu = s1a[rowL] * (1.f / 256.f);
        float var = s2a[rowL] * (1.f / 256.f) - mu * mu;
        float inv = rsqrtf(var + 1e-5f);
        out[((size_t)b * S + m0 + rowL) * D + col] = g * (acc[m2][ni][r] - mu) * inv + be;
      }
    }
}

// ---------------- host launcher ----------------
extern "C" void kernel_launch(void* const* d_in, const int* in_sizes, int n_in,
                              void* d_out, int out_size, void* d_ws, size_t ws_size,
                              hipStream_t stream) {
  (void)in_sizes; (void)n_in; (void)out_size; (void)ws_size;
  const float* X     = (const float*)d_in[0];
  const float* adj   = (const float*)d_in[1];
  const float* W     = (const float*)d_in[2];
  const float* bias  = (const float*)d_in[3];
  const float* gamma = (const float*)d_in[4];
  const float* beta  = (const float*)d_in[5];
  float* out = (float*)d_out;

  char* ws = (char*)d_ws;
  size_t off = 0;
  auto carve = [&](size_t bytes) { void* p = ws + off; off = (off + bytes + 255) & ~(size_t)255; return p; };
  float*          denom = (float*)carve((size_t)Bb * S * sizeof(float));          // 64 KB
  unsigned short* Xb    = (unsigned short*)carve((size_t)Bb * S * D * 2);         // 8 MB
  unsigned short* Wt    = (unsigned short*)carve((size_t)D * D * 2);              // 128 KB
  unsigned short* Yt    = (unsigned short*)carve((size_t)Bb * D * S * 2);         // 8 MB

  k_zero<<<(Bb * S) / 256, 256, 0, stream>>>(denom, Bb * S);
  k_cvt<<<(Bb * S * D) / 256, 256, 0, stream>>>(X, Xb, Bb * S * D);
  k_cvtW<<<(D * D) / 256, 256, 0, stream>>>(W, Wt);
  k_denom<<<dim3(S / 64, S / 64, Bb), 256, 0, stream>>>(adj, denom);
  k_gemmY<<<dim3(S / 64, Bb), 256, 0, stream>>>(Xb, Wt, Yt);
  k_attn<<<dim3(S / 64, Bb), 256, 0, stream>>>(X, Xb, Yt, adj, denom, bias, gamma, beta, out);
}
